// QuantizedLinear_34634616275187
// MI455X (gfx1250) — compile-verified, matrix operand order checked by reading, not run on hardware
//
#include <hip/hip_runtime.h>
#include <cstdint>
#include <cstddef>

// ---------------------------------------------------------------------------
// QuantizedLinear: out[M,N] = scale * (A[M,K] x (Q - zp)^T[K,N]) + bias[N]
//   M = 4*2048 = 8192, K = 1024, N = 50257
// Two-pass: pre-convert A -> bf16, W=(q-zp) -> bf16 into d_ws (120 MB, fits
// MI455X 192MB L2), then bf16 WMMA GEMM with async LDS staging.
// Main GEMM: block tile 256x128, 8 waves, 64x64 per wave (4x4 WMMAs),
// double-buffered global_load_async_to_lds_b128, one barrier per K-step.
// ---------------------------------------------------------------------------

#define MTOT 8192
#define KDIM 1024
#define NTOT 50257
#define BM 256   // main GEMM M tile
#define BN 128   // main GEMM N tile
#define BK 32
#define LDSK 40 // padded LDS k-stride (elements): 80B rows, bank-conflict free
#define BMF 128  // fused-fallback M tile

typedef __bf16 bf16_t;
typedef __attribute__((ext_vector_type(16))) __bf16 v16bf;
typedef __attribute__((ext_vector_type(8)))  __bf16 v8bf;
typedef __attribute__((ext_vector_type(8)))  float  v8f;

#if defined(__HIP_DEVICE_COMPILE__) && __has_builtin(__builtin_amdgcn_global_load_async_to_lds_b128)
#define HAVE_ASYNC_LDS 1
#else
#define HAVE_ASYNC_LDS 0
#endif

#if defined(__HIP_DEVICE_COMPILE__)
typedef int i32x4 __attribute__((vector_size(16)));
typedef __attribute__((address_space(1))) i32x4 g1_i32x4;  // global-AS int4
typedef __attribute__((address_space(3))) i32x4 lds_i32x4; // LDS-AS int4
#endif

static __device__ __forceinline__ void cp16_to_lds(const void* g, void* l) {
#if HAVE_ASYNC_LDS
  // CDNA5 async copy: global -> LDS, tracked by ASYNCcnt.
  __builtin_amdgcn_global_load_async_to_lds_b128(
      (g1_i32x4*)(uintptr_t)g,
      (lds_i32x4*)(uint32_t)(uintptr_t)l, // low 32 bits of flat addr == LDS byte addr
      0, 0);
#else
  *(uint4*)l = *(const uint4*)g;
#endif
}

static __device__ __forceinline__ void wait_async_copies() {
#if HAVE_ASYNC_LDS
#if __has_builtin(__builtin_amdgcn_s_wait_asynccnt)
  __builtin_amdgcn_s_wait_asynccnt(0);
#else
  asm volatile("s_wait_asynccnt 0" ::: "memory");
#endif
#endif
}

static __device__ __forceinline__ unsigned short f2bf(float f) {
  union { float f; uint32_t u; } x;
  x.f = f;
  x.u += 0x7FFFu + ((x.u >> 16) & 1u); // round-to-nearest-even
  return (unsigned short)(x.u >> 16);
}

// ---------------------------------------------------------------------------
// WMMA compute over a MI*16 x NI*16 wave sub-tile.
// 16-bit fragment per ISA layout: lanes 0-15 hold K 0-7 & 16-23,
// lanes 16-31 hold K 8-15 & 24-31 -> two 16B ds loads per fragment.
// B fragments are loaded once and reused across all MI row-tiles.
// ---------------------------------------------------------------------------
union FragU { v16bf v; v8bf h[2]; };

template <int MI, int NI>
static __device__ __forceinline__ void mma_tiles(const bf16_t* As, const bf16_t* Ws,
                                                 int lane, int mbase, int nbase,
                                                 v8f (&c)[MI][NI]) {
  const int rsel = lane & 15;
  const int klo  = (lane >> 4) << 3; // 0 or 8
  FragU b[NI];
#pragma unroll
  for (int ni = 0; ni < NI; ++ni) {
    const bf16_t* p = Ws + (nbase + ni * 16 + rsel) * LDSK + klo;
    b[ni].h[0] = *(const v8bf*)(p);
    b[ni].h[1] = *(const v8bf*)(p + 16);
  }
#pragma unroll
  for (int mi = 0; mi < MI; ++mi) {
    FragU a;
    const bf16_t* p = As + (mbase + mi * 16 + rsel) * LDSK + klo;
    a.h[0] = *(const v8bf*)(p);
    a.h[1] = *(const v8bf*)(p + 16);
#pragma unroll
    for (int ni = 0; ni < NI; ++ni)
      c[mi][ni] = __builtin_amdgcn_wmma_f32_16x16x32_bf16(
          false, a.v, false, b[ni].v, (short)0, c[mi][ni], false, false);
  }
}

template <int MI, int NI>
static __device__ __forceinline__ void epilogue(v8f (&c)[MI][NI], const float* bias,
                                                float s, float* out,
                                                int mb, int nb, int lane) {
  float bn[NI];
#pragma unroll
  for (int ni = 0; ni < NI; ++ni) {
    int n = nb + ni * 16 + (lane & 15);
    bn[ni] = (n < NTOT) ? bias[n] : 0.f;
  }
#pragma unroll
  for (int mi = 0; mi < MI; ++mi) {
    const int mrow = mb + mi * 16 + ((lane >> 4) << 3);
#pragma unroll
    for (int ni = 0; ni < NI; ++ni) {
      const int n = nb + ni * 16 + (lane & 15);
      if (n < NTOT) {
#pragma unroll
        for (int i = 0; i < 8; ++i)
          out[(size_t)(mrow + i) * NTOT + n] = c[mi][ni][i] * s + bn[ni];
      }
    }
  }
}

// ---------------------------------------------------------------------------
// Pre-pass 1: f32 -> bf16 (A)
// ---------------------------------------------------------------------------
__global__ __launch_bounds__(256) void cvt_in_kernel(const float* __restrict__ in,
                                                     bf16_t* __restrict__ o, long n) {
  long i = ((long)blockIdx.x * 256 + threadIdx.x) * 8;
  if (i >= n) return;
  float4 a = *(const float4*)(in + i);
  float4 b = *(const float4*)(in + i + 4);
  float t[8] = {a.x, a.y, a.z, a.w, b.x, b.y, b.z, b.w};
  union { unsigned short s[8]; uint4 v; } pk;
#pragma unroll
  for (int j = 0; j < 8; ++j) pk.s[j] = f2bf(t[j]);
  *(uint4*)(o + i) = pk.v;
}

// ---------------------------------------------------------------------------
// Pre-pass 2: (q - zp) -> bf16 (W). Values in [-135,135] are exact in bf16;
// scale is folded into the GEMM epilogue.
// ---------------------------------------------------------------------------
__global__ __launch_bounds__(256) void dq_w_kernel(const int* __restrict__ q,
                                                   const int* __restrict__ zp,
                                                   bf16_t* __restrict__ o, long n) {
  long i = ((long)blockIdx.x * 256 + threadIdx.x) * 8;
  if (i >= n) return;
  const int z = zp[0];
  int4 a = *(const int4*)(q + i);
  int4 b = *(const int4*)(q + i + 4);
  int t[8] = {a.x, a.y, a.z, a.w, b.x, b.y, b.z, b.w};
  union { unsigned short s[8]; uint4 v; } pk;
#pragma unroll
  for (int j = 0; j < 8; ++j) pk.s[j] = f2bf((float)(t[j] - z));
  *(uint4*)(o + i) = pk.v;
}

// ---------------------------------------------------------------------------
// Main GEMM: 256x128 block, 8 waves (4m x 2n), 64x64 per wave.
// Double-buffered async LDS staging; single barrier per K-step: the barrier
// after s_wait_asynccnt also proves all waves finished reading the other
// buffer in the previous iteration, so stage() may overwrite it.
// ---------------------------------------------------------------------------
__global__ __launch_bounds__(256) void qlin_gemm_bf16(const bf16_t* __restrict__ Abf,
                                                      const bf16_t* __restrict__ Wd,
                                                      const float* __restrict__ scale,
                                                      const float* __restrict__ bias,
                                                      float* __restrict__ out) {
  __shared__ alignas(16) bf16_t As[2][BM * LDSK]; // 2 x 20 KB
  __shared__ alignas(16) bf16_t Ws[2][BN * LDSK]; // 2 x 10 KB

  const int t = threadIdx.x;
  const int lane = t & 31, wave = t >> 5;
  const int wm = wave >> 1, wn = wave & 1; // 4 x 2 wave grid
  const int m0 = blockIdx.y * BM;
  const int n0 = blockIdx.x * BN;

  v8f c[4][4];
#pragma unroll
  for (int mi = 0; mi < 4; ++mi)
#pragma unroll
    for (int ni = 0; ni < 4; ++ni)
      c[mi][ni] = (v8f){0.f, 0.f, 0.f, 0.f, 0.f, 0.f, 0.f, 0.f};

  // A: 1024 16-byte chunks (4/thread); W: 512 chunks (2/thread).
  auto stage = [&](int kt, int buf) {
    const int kbase = kt * BK;
#pragma unroll
    for (int j = 0; j < 4; ++j) {
      const int ch = t + (j << 8);
      const int row = ch >> 2;
      const int ko = (ch & 3) << 3;
      cp16_to_lds(Abf + (size_t)(m0 + row) * KDIM + kbase + ko,
                  &As[buf][row * LDSK + ko]);
    }
#pragma unroll
    for (int j = 0; j < 2; ++j) {
      const int ch = t + (j << 8);
      const int row = ch >> 2;
      const int ko = (ch & 3) << 3;
      int wr = n0 + row;
      wr = wr < NTOT ? wr : NTOT - 1; // clamp edge rows (stores are guarded)
      cp16_to_lds(Wd + (size_t)wr * KDIM + kbase + ko,
                  &Ws[buf][row * LDSK + ko]);
    }
  };

  const int NKI = KDIM / BK; // 32 K-steps
  stage(0, 0);
  int buf = 0;
  for (int kt = 0; kt < NKI; ++kt) {
    wait_async_copies(); // my wave's copies into `buf` complete
    __syncthreads();     // all copies visible; all waves done reading buf^1
    if (kt + 1 < NKI) stage(kt + 1, buf ^ 1); // overlap next stage w/ compute
    mma_tiles<4, 4>(As[buf], Ws[buf], lane, wm * 64, wn * 64, c);
    buf ^= 1;
  }

  epilogue<4, 4>(c, bias, scale[0], out, m0 + wm * 64, n0 + wn * 64, lane);
}

// ---------------------------------------------------------------------------
// Fallback fused GEMM (only if d_ws is too small for bf16 operands):
// dequantize / downconvert in the staging path, single LDS buffer,
// 128x128 block, 32x64 per wave.
// ---------------------------------------------------------------------------
__global__ __launch_bounds__(256) void qlin_gemm_fused(const float* __restrict__ A,
                                                       const int* __restrict__ Q,
                                                       const float* __restrict__ scale,
                                                       const int* __restrict__ zp,
                                                       const float* __restrict__ bias,
                                                       float* __restrict__ out) {
  __shared__ alignas(16) bf16_t As[BMF * LDSK];
  __shared__ alignas(16) bf16_t Ws[BN * LDSK];

  const int t = threadIdx.x;
  const int lane = t & 31, wave = t >> 5;
  const int wm = wave >> 1, wn = wave & 1;
  const int m0 = blockIdx.y * BMF;
  const int n0 = blockIdx.x * BN;
  const int z = zp[0];

  const int rowA = t >> 1;     // 2 threads per 32-wide row
  const int kh = (t & 1) << 4; // 0 or 16

  v8f c[2][4];
#pragma unroll
  for (int mi = 0; mi < 2; ++mi)
#pragma unroll
    for (int ni = 0; ni < 4; ++ni)
      c[mi][ni] = (v8f){0.f, 0.f, 0.f, 0.f, 0.f, 0.f, 0.f, 0.f};

  float4 ra[4];
  int4 rq[4];
  auto gload = [&](int kt) {
    const float* pa = A + (size_t)(m0 + rowA) * KDIM + kt * BK + kh;
#pragma unroll
    for (int i = 0; i < 4; ++i) ra[i] = ((const float4*)pa)[i];
    int wr = n0 + rowA;
    wr = wr < NTOT ? wr : NTOT - 1;
    const int* pq = Q + (size_t)wr * KDIM + kt * BK + kh;
#pragma unroll
    for (int i = 0; i < 4; ++i) rq[i] = ((const int4*)pq)[i];
  };
  auto commit = [&]() {
    union { unsigned short s[8]; uint4 v; } pk;
#pragma unroll
    for (int h = 0; h < 2; ++h) {
      const float* f = (const float*)&ra[h * 2];
#pragma unroll
      for (int i = 0; i < 8; ++i) pk.s[i] = f2bf(f[i]);
      *(uint4*)&As[rowA * LDSK + kh + h * 8] = pk.v;
    }
#pragma unroll
    for (int h = 0; h < 2; ++h) {
      const int* qq = (const int*)&rq[h * 2];
#pragma unroll
      for (int i = 0; i < 8; ++i) pk.s[i] = f2bf((float)(qq[i] - z));
      *(uint4*)&Ws[rowA * LDSK + kh + h * 8] = pk.v;
    }
  };

  const int NKI = KDIM / BK;
  gload(0);
  for (int kt = 0; kt < NKI; ++kt) {
    __syncthreads(); // previous compute done before overwriting LDS
    commit();
    __syncthreads();
    if (kt + 1 < NKI) gload(kt + 1); // prefetch next tile into regs
    mma_tiles<2, 4>(As, Ws, lane, wm * 32, wn * 64, c);
  }

  epilogue<2, 4>(c, bias, scale[0], out, m0 + wm * 32, n0 + wn * 64, lane);
}

// ---------------------------------------------------------------------------
extern "C" void kernel_launch(void* const* d_in, const int* in_sizes, int n_in,
                              void* d_out, int out_size, void* d_ws, size_t ws_size,
                              hipStream_t stream) {
  (void)in_sizes; (void)n_in; (void)out_size;
  const float* inp  = (const float*)d_in[0];
  const int*  qw    = (const int*)d_in[1];
  const float* scl  = (const float*)d_in[2];
  const int*  zp    = (const int*)d_in[3];
  const float* bias = (const float*)d_in[4];
  float* out = (float*)d_out;

  const size_t needA = (size_t)MTOT * KDIM * sizeof(unsigned short); //  16.8 MB
  const size_t needW = (size_t)NTOT * KDIM * sizeof(unsigned short); // 102.9 MB

  dim3 block(256);

  if (ws_size >= needA + needW) {
    bf16_t* Abf = (bf16_t*)d_ws;
    bf16_t* Wdq = (bf16_t*)((char*)d_ws + needA);
    const long nA = (long)MTOT * KDIM;
    const long nW = (long)NTOT * KDIM;
    cvt_in_kernel<<<(unsigned)(nA / 8 / 256), block, 0, stream>>>(inp, Abf, nA);
    dq_w_kernel<<<(unsigned)((nW / 8 + 255) / 256), block, 0, stream>>>(qw, zp, Wdq, nW);
    dim3 grid((NTOT + BN - 1) / BN, MTOT / BM); // 393 x 32 workgroups
    qlin_gemm_bf16<<<grid, block, 0, stream>>>(Abf, Wdq, scl, bias, out);
  } else {
    dim3 grid((NTOT + BN - 1) / BN, MTOT / BMF); // 393 x 64 workgroups
    qlin_gemm_fused<<<grid, block, 0, stream>>>(inp, qw, scl, zp, bias, out);
  }
}
